// RetinaNet_D2_1537598292575
// MI455X (gfx1250) — matrix operand order repaired and checked by reading, not compiled, not run on hardware
//
#include <hip/hip_runtime.h>
#include <hip/hip_bf16.h>
#include <math.h>

// ---------------------------------------------------------------------------
// RetinaNet-D2 head on gfx1250 (MI455X).
// Convs = implicit GEMM via v_wmma_f32_16x16x32_bf16 (f32 accum), NHWC bf16
// activations, weights repacked+zero-padded [co][tap][ci] bf16.
// 64x64 block tile, 8 waves x (16x32). K chunked per-tap (64x128 tiles):
// double-buffered GLOBAL_LOAD_ASYNC_TO_LDS_B128 staging (in-order ASYNCcnt
// lets us overlap chunk c+1's loads with chunk c's 8 WMMAs), border pixels
// redirected to a zero guard buffer.
// ---------------------------------------------------------------------------

typedef __attribute__((ext_vector_type(16))) __bf16 v16bf;
typedef __attribute__((ext_vector_type(8)))  float  v8f;

#define C_IN        256
#define KTOT        2304          // 9 * 256
#define A_ANCH      9
#define KCLS        80
#define N_IMG       2
#define TOT_CAND    4630
#define MAX_DET_    100
#define NMS_TH_     0.5f
#define SCORE_TH_   0.05f
#define IMG_W_      1216.0f
#define IMG_H_      800.0f
#define SCALE_CLAMP_ 4.135166556742356f   // log(1000/16)

// ===== weight repack: (R,256,3,3) f32 -> [r][tap][ci] bf16, zero-padded rows
__global__ void repack_w(const float* __restrict__ src, unsigned short* __restrict__ dst,
                         int rows_real, int rows_pad) {
    __bf16* d = (__bf16*)dst;
    size_t total = (size_t)rows_pad * KTOT;
    for (size_t i = (size_t)blockIdx.x * blockDim.x + threadIdx.x; i < total;
         i += (size_t)gridDim.x * blockDim.x) {
        int ci  = (int)(i % 256);
        int tap = (int)((i / 256) % 9);
        int row = (int)(i / KTOT);
        float v = (row < rows_real) ? src[((size_t)row * 256 + ci) * 9 + tap] : 0.f;
        d[i] = (__bf16)v;
    }
}

// ===== features: NCHW f32 -> NHWC bf16
__global__ void nchw2nhwc(const float* __restrict__ src, unsigned short* __restrict__ dst,
                          int HW) {
    __bf16* d = (__bf16*)dst;
    size_t total = (size_t)N_IMG * HW * C_IN;
    for (size_t i = (size_t)blockIdx.x * blockDim.x + threadIdx.x; i < total;
         i += (size_t)gridDim.x * blockDim.x) {
        int c = (int)(i % C_IN);
        size_t t = i / C_IN;
        int p = (int)(t % HW);
        int n = (int)(t / HW);
        d[i] = (__bf16)src[((size_t)n * C_IN + c) * HW + p];
    }
}

// ===== small zero-fill (guard buffer for border pixels) ====================
__global__ void zero_u32(unsigned int* p, int n) {
    int i = blockIdx.x * blockDim.x + threadIdx.x;
    if (i < n) p[i] = 0u;
}

// ===== conv3x3 implicit GEMM with WMMA bf16 ================================
// block: 256 thr = 8 waves, tile 64 cout x 64 px; wave = 16 cout x 32 px.
// LDS: 2x (A 64x128 + B 64x128) bf16 tiles (row stride 272B), async staged.
#define ROWB  272                 // 128 bf16 = 256B data + 16B pad
#define TILEB (64 * ROWB)         // 17408 B per tile
template <bool RELU, bool OUT_BF16>
__global__ __launch_bounds__(256)
void conv3_wmma(const unsigned short* __restrict__ in_u,   // NHWC bf16 [n][p][256]
                const unsigned short* __restrict__ wt_u,   // bf16 [co_pad][2304]
                const float* __restrict__ bias,            // [Cout]
                const unsigned short* __restrict__ zguard, // 512B of zeros
                void* __restrict__ outp,
                int H, int W, int Cout) {
    const __bf16* in = (const __bf16*)in_u;
    const __bf16* wt = (const __bf16*)wt_u;
    const int HW  = H * W;
    const int n   = blockIdx.z;
    const int p0  = blockIdx.x * 64;
    const int co0 = blockIdx.y * 64;
    const int tid = threadIdx.x;

    __shared__ __align__(16) unsigned char smem[4 * TILEB]; // A0 A1 B0 B1
    unsigned char* ldsA = smem;                 // + buf*TILEB
    unsigned char* ldsB = smem + 2 * TILEB;     // + buf*TILEB

    const int lane = tid & 31;
    const int wave = tid >> 5;
    const int wm = wave & 3;   // M tile 0..3
    const int wn = wave >> 2;  // N tile 0..1 (32 px each)

    v8f acc0 = {0.f, 0.f, 0.f, 0.f, 0.f, 0.f, 0.f, 0.f};
    v8f acc1 = {0.f, 0.f, 0.f, 0.f, 0.f, 0.f, 0.f, 0.f};

    // staging coords: thread owns row am, 16B chunk ch; 4 async b128 x 64B span
    const int am = tid >> 2;   // 0..63
    const int ch = tid & 3;

    int p = p0 + am;
    const int pvalid = (p < HW);
    int pc = pvalid ? p : (HW - 1);
    const int py  = pc / W;
    const int pxc = pc - py * W;

    const __bf16* wrow   = wt + (size_t)(co0 + am) * KTOT;   // rows zero-padded
    const __bf16* inbase = in + (size_t)n * HW * C_IN;
    const __bf16* center = inbase + ((size_t)py * W + pxc) * C_IN;

    // per-thread LDS dest addresses (wave-relative, low 32 bits of flat addr)
    const unsigned int lA = (unsigned int)(uintptr_t)(ldsA + am * ROWB + ch * 16);
    const unsigned int lB = (unsigned int)(uintptr_t)(ldsB + am * ROWB + ch * 16);

    // fragment base addresses (A 16x32 bf16, B 32x16 bf16; lane half h holds contiguous k 16h..16h+15)
    const unsigned char* fA =
        ldsA + (wm * 16 + (lane & 15)) * ROWB + ((lane < 16) ? 0 : 32);
    const unsigned char* fB0 =
        ldsB + (wn * 32 + (lane & 15)) * ROWB + ((lane < 16) ? 0 : 32);
    const unsigned char* fB1 = fB0 + 16 * ROWB;

    // issue one 64x128 A-tile + B-tile as 8 async direct-to-LDS b128 loads
    auto issue = [&](int c, int buf) {
        const int tap  = c >> 1;
        const int half = (c & 1) << 7;      // ci offset 0 / 128
        const int dy = tap / 3 - 1, dx = tap - (tap / 3) * 3 - 1;
        int yy = py + dy, xx = pxc + dx;
        int ok = pvalid & (yy >= 0) & (yy < H) & (xx >= 0) & (xx < W);
        // invalid border lanes fetch from the zero guard buffer instead
        const __bf16* bp = ok ? (center + ((size_t)dy * W + dx) * C_IN + half)
                              : ((const __bf16*)zguard);
        unsigned long long ga =
            (unsigned long long)(uintptr_t)(wrow + (size_t)tap * 256 + half + ch * 8);
        unsigned long long gb = (unsigned long long)(uintptr_t)(bp + ch * 8);
        unsigned int la = lA + buf * TILEB;
        unsigned int lb = lB + buf * TILEB;
        // offset: applies to BOTH the LDS and global address (ISA 15.18)
        asm volatile(
            "global_load_async_to_lds_b128 %0, %2, off\n\t"
            "global_load_async_to_lds_b128 %0, %2, off offset:64\n\t"
            "global_load_async_to_lds_b128 %0, %2, off offset:128\n\t"
            "global_load_async_to_lds_b128 %0, %2, off offset:192\n\t"
            "global_load_async_to_lds_b128 %1, %3, off\n\t"
            "global_load_async_to_lds_b128 %1, %3, off offset:64\n\t"
            "global_load_async_to_lds_b128 %1, %3, off offset:128\n\t"
            "global_load_async_to_lds_b128 %1, %3, off offset:192"
            :: "v"(la), "v"(lb), "v"(ga), "v"(gb) : "memory");
    };

    issue(0, 0);
    for (int c = 0; c < 18; ++c) {          // 9 taps x 2 half-K chunks
        const int buf = c & 1;
        if (c + 1 < 18) {
            issue(c + 1, buf ^ 1);          // next chunk in flight during compute
            // async loads complete in order: <=8 outstanding => chunk c landed
            asm volatile("s_wait_asynccnt 0x8" ::: "memory");
        } else {
            asm volatile("s_wait_asynccnt 0x0" ::: "memory");
        }
        __syncthreads();                     // publish chunk c to all waves

        const unsigned char* A  = fA  + buf * TILEB;
        const unsigned char* B0 = fB0 + buf * TILEB;
        const unsigned char* B1 = fB1 + buf * TILEB;
        // ---- 4 unrolled K-steps of 32, all LDS reads at immediate offsets
#pragma unroll
        for (int s = 0; s < 4; ++s) {
            union { uint4 q[2]; v16bf v; } af, bm0, bm1;
            af.q[0]  = *(const uint4*)(A + s * 64);
            af.q[1]  = *(const uint4*)(A + s * 64 + 16);
            bm0.q[0] = *(const uint4*)(B0 + s * 64);
            bm0.q[1] = *(const uint4*)(B0 + s * 64 + 16);
            bm1.q[0] = *(const uint4*)(B1 + s * 64);
            bm1.q[1] = *(const uint4*)(B1 + s * 64 + 16);
            acc0 = __builtin_amdgcn_wmma_f32_16x16x32_bf16(false, af.v, false, bm0.v,
                                                           (short)0, acc0, false, false);
            acc1 = __builtin_amdgcn_wmma_f32_16x16x32_bf16(false, af.v, false, bm1.v,
                                                           (short)0, acc1, false, false);
        }
        __syncthreads();                     // reads done before buf is re-staged
    }

    // ---- epilogue: lane owns 8 consecutive cout at 2 pixels
    const int cob = co0 + wm * 16 + ((lane < 16) ? 0 : 8);
    const int pg0 = p0 + wn * 32 + (lane & 15);
    const int pg1 = pg0 + 16;
    float bv[8];
#pragma unroll
    for (int v = 0; v < 8; ++v) bv[v] = (cob + v < Cout) ? bias[cob + v] : 0.f;

    if (OUT_BF16) {  // Cout == 256 -> full 8-group, vectorized b128 stores
        union { __bf16 h[8]; uint4 q; } pk0, pk1;
#pragma unroll
        for (int v = 0; v < 8; ++v) {
            float x0 = acc0[v] + bv[v];
            float x1 = acc1[v] + bv[v];
            if (RELU) { x0 = fmaxf(x0, 0.f); x1 = fmaxf(x1, 0.f); }
            pk0.h[v] = (__bf16)x0;
            pk1.h[v] = (__bf16)x1;
        }
        if (pg0 < HW)
            *(uint4*)((__bf16*)outp + ((size_t)n * HW + pg0) * C_IN + cob) = pk0.q;
        if (pg1 < HW)
            *(uint4*)((__bf16*)outp + ((size_t)n * HW + pg1) * C_IN + cob) = pk1.q;
    } else {         // f32 NHWC out (heads), bounds-checked (Cout 720 / 36)
#pragma unroll
        for (int v = 0; v < 8; ++v) {
            int co = cob + v;
            if (co < Cout) {
                float x0 = acc0[v] + bv[v];
                float x1 = acc1[v] + bv[v];
                if (RELU) { x0 = fmaxf(x0, 0.f); x1 = fmaxf(x1, 0.f); }
                if (pg0 < HW) ((float*)outp)[((size_t)n * HW + pg0) * Cout + co] = x0;
                if (pg1 < HW) ((float*)outp)[((size_t)n * HW + pg1) * Cout + co] = x1;
            }
        }
    }
}

// ======================= top-k (radix select on float keys) ================
__device__ __forceinline__ unsigned int f2key(float f) {
    unsigned int b = __float_as_uint(f);
    return (b & 0x80000000u) ? ~b : (b | 0x80000000u);
}

__global__ void topk_init(unsigned int* hist, unsigned int* state, int k) {
    int i = blockIdx.x * blockDim.x + threadIdx.x;
    if (i < N_IMG * 256) hist[i] = 0u;
    if (i < N_IMG) {
        state[i * 4 + 0] = 0u;           // prefix
        state[i * 4 + 1] = (unsigned)k;  // kleft
        state[i * 4 + 2] = 0u;           // cnt >
        state[i * 4 + 3] = 0u;           // cnt ==
    }
}

__global__ void topk_hist(const float* __restrict__ logits, int M,
                          unsigned int* __restrict__ hist,
                          const unsigned int* __restrict__ state,
                          int pass, int shift) {
    int n = blockIdx.z;
    const float* src = logits + (size_t)n * M;
    unsigned int prefix = state[n * 4 + 0];
    __shared__ unsigned int lh[256];
    lh[threadIdx.x] = 0u;
    __syncthreads();
    for (int e = blockIdx.x * blockDim.x + threadIdx.x; e < M;
         e += gridDim.x * blockDim.x) {
        unsigned int key = f2key(src[e]);
        bool ok = (pass == 0) || (((key ^ prefix) >> (shift + 8)) == 0u);
        if (ok) atomicAdd(&lh[(key >> shift) & 255u], 1u);
    }
    __syncthreads();
    if (lh[threadIdx.x]) atomicAdd(&hist[n * 256 + threadIdx.x], lh[threadIdx.x]);
}

__global__ void topk_pick(unsigned int* hist, unsigned int* state, int shift) {
    int n = blockIdx.x;
    __shared__ unsigned int lh[256];
    lh[threadIdx.x] = hist[n * 256 + threadIdx.x];
    hist[n * 256 + threadIdx.x] = 0u;  // re-zero for next pass
    __syncthreads();
    if (threadIdx.x == 0) {
        unsigned int kleft = state[n * 4 + 1];
        unsigned int cum = 0u;
        int b;
        for (b = 255; b > 0; --b) {
            if (cum + lh[b] >= kleft) break;
            cum += lh[b];
        }
        state[n * 4 + 0] |= ((unsigned)b) << shift;
        state[n * 4 + 1] = kleft - cum;  // remaining within chosen digit
    }
}

__global__ void topk_compact(const float* __restrict__ logits, int M,
                             unsigned int* state, int* tmpA, int* tmpC,
                             float* tmpL, int k) {
    int n = blockIdx.z;
    const float* src = logits + (size_t)n * M;
    unsigned int T = state[n * 4 + 0];
    unsigned int kleft = state[n * 4 + 1];
    int base_eq = k - (int)kleft;
    for (int e = blockIdx.x * blockDim.x + threadIdx.x; e < M;
         e += gridDim.x * blockDim.x) {
        float lv = src[e];
        unsigned int key = f2key(lv);
        int slot = -1;
        if (key > T) {
            slot = (int)atomicAdd(&state[n * 4 + 2], 1u);
        } else if (key == T) {
            unsigned int j = atomicAdd(&state[n * 4 + 3], 1u);
            if (j < kleft) slot = base_eq + (int)j;
        }
        if (slot >= 0 && slot < k) {
            tmpA[n * 1000 + slot] = e / KCLS;
            tmpC[n * 1000 + slot] = e % KCLS;
            tmpL[n * 1000 + slot] = lv;
        }
    }
}

// ======================= decode + clip + score threshold ===================
__global__ void decode_kernel(const int* __restrict__ tmpA, const int* __restrict__ tmpC,
                              const float* __restrict__ tmpL,
                              const float* __restrict__ reg,   // NHWC f32 (n,HW,36)
                              const float* __restrict__ anc,   // (HWA,4)
                              float* candB, float* candS, int* candC,
                              int k, int lvl_off, int HW) {
    int n = blockIdx.z;
    int s = blockIdx.x * blockDim.x + threadIdx.x;
    if (s >= k) return;
    int ai = tmpA[n * 1000 + s];
    int cl = tmpC[n * 1000 + s];
    float lg = tmpL[n * 1000 + s];
    int p = ai / A_ANCH, a = ai - p * A_ANCH;
    const float* d = reg + ((size_t)n * HW + p) * 36 + a * 4;
    const float* an = anc + (size_t)ai * 4;
    float wa = an[2] - an[0], ha = an[3] - an[1];
    float xa = an[0] + 0.5f * wa, ya = an[1] + 0.5f * ha;
    float dw = fminf(d[2], SCALE_CLAMP_), dh = fminf(d[3], SCALE_CLAMP_);
    float cx = d[0] * wa + xa, cy = d[1] * ha + ya;
    float w = __expf(dw) * wa, h = __expf(dh) * ha;
    float x0 = fminf(fmaxf(cx - 0.5f * w, 0.f), IMG_W_);
    float y0 = fminf(fmaxf(cy - 0.5f * h, 0.f), IMG_H_);
    float x1 = fminf(fmaxf(cx + 0.5f * w, 0.f), IMG_W_);
    float y1 = fminf(fmaxf(cy + 0.5f * h, 0.f), IMG_H_);
    float prob = 1.f / (1.f + __expf(-lg));
    float sc = (prob > SCORE_TH_) ? prob : 0.f;
    size_t o = (size_t)n * TOT_CAND + lvl_off + s;
    candB[o * 4 + 0] = x0; candB[o * 4 + 1] = y0;
    candB[o * 4 + 2] = x1; candB[o * 4 + 3] = y1;
    candS[o] = sc;
    candC[o] = cl;
}

// ======================= class-aware NMS (1 block / image) =================
__global__ __launch_bounds__(256)
void nms_kernel(const float* __restrict__ candB, const float* __restrict__ candS,
                const int* __restrict__ candC, float* __restrict__ out) {
    int n = blockIdx.x;
    __shared__ float sS[TOT_CAND];
    __shared__ float rv[256];
    __shared__ int ri[256];
    __shared__ int sBI[MAX_DET_];
    __shared__ float sBS[MAX_DET_];
    __shared__ float jb[4];
    const float* B = candB + (size_t)n * TOT_CAND * 4;
    const float* S = candS + (size_t)n * TOT_CAND;
    const int* Cl = candC + (size_t)n * TOT_CAND;

    for (int e = threadIdx.x; e < TOT_CAND; e += 256) sS[e] = S[e];
    __syncthreads();

    for (int it = 0; it < MAX_DET_; ++it) {
        float bvv = -1.f; int bi = 0;
        for (int e = threadIdx.x; e < TOT_CAND; e += 256) {
            float v = sS[e];
            if (v > bvv) { bvv = v; bi = e; }
        }
        rv[threadIdx.x] = bvv; ri[threadIdx.x] = bi;
        __syncthreads();
        for (int s = 128; s > 0; s >>= 1) {
            if (threadIdx.x < s) {
                float v2 = rv[threadIdx.x + s];
                if (v2 > rv[threadIdx.x] ||
                    (v2 == rv[threadIdx.x] && ri[threadIdx.x + s] < ri[threadIdx.x])) {
                    rv[threadIdx.x] = v2; ri[threadIdx.x] = ri[threadIdx.x + s];
                }
            }
            __syncthreads();
        }
        int j = ri[0];
        if (threadIdx.x == 0) {
            sBI[it] = j; sBS[it] = rv[0];
            float off = (float)Cl[j] * 4096.f;
            jb[0] = B[j * 4 + 0] + off; jb[1] = B[j * 4 + 1] + off;
            jb[2] = B[j * 4 + 2] + off; jb[3] = B[j * 4 + 3] + off;
            sS[j] = 0.f;
        }
        __syncthreads();
        float ax0 = jb[0], ay0 = jb[1], ax1 = jb[2], ay1 = jb[3];
        float aarea = (ax1 - ax0) * (ay1 - ay0);
        for (int e = threadIdx.x; e < TOT_CAND; e += 256) {
            float off = (float)Cl[e] * 4096.f;
            float bx0 = B[e * 4 + 0] + off, by0 = B[e * 4 + 1] + off;
            float bx1 = B[e * 4 + 2] + off, by1 = B[e * 4 + 3] + off;
            float iw = fmaxf(fminf(ax1, bx1) - fmaxf(ax0, bx0), 0.f);
            float ih = fmaxf(fminf(ay1, by1) - fmaxf(ay0, by0), 0.f);
            float inter = iw * ih;
            float barea = (bx1 - bx0) * (by1 - by0);
            float iou = inter / (aarea + barea - inter + 1e-9f);
            if (iou > NMS_TH_) sS[e] = 0.f;
        }
        __syncthreads();
    }
    // outputs: boxes[2,100,4] | scores[2,100] | classes[2,100]
    for (int i = threadIdx.x; i < MAX_DET_; i += 256) {
        int j = sBI[i];
        float* ob = out + ((size_t)n * MAX_DET_ + i) * 4;
        ob[0] = B[j * 4 + 0]; ob[1] = B[j * 4 + 1];
        ob[2] = B[j * 4 + 2]; ob[3] = B[j * 4 + 3];
        out[N_IMG * MAX_DET_ * 4 + n * MAX_DET_ + i] = sBS[i];
        out[N_IMG * MAX_DET_ * 5 + n * MAX_DET_ + i] = (float)Cl[j];
    }
}

// ======================= host orchestration ================================
extern "C" void kernel_launch(void* const* d_in, const int* in_sizes, int n_in,
                              void* d_out, int out_size, void* d_ws, size_t ws_size,
                              hipStream_t stream) {
    (void)in_sizes; (void)n_in; (void)out_size; (void)ws_size;
    static const int Hs[5] = {100, 50, 25, 13, 7};
    static const int Ws[5] = {152, 76, 38, 19, 10};
    static const int NTK[5] = {1000, 1000, 1000, 1000, 630};
    static const int LOFF[5] = {0, 1000, 2000, 3000, 4000};

    const float* feats[5];
    for (int i = 0; i < 5; ++i) feats[i] = (const float*)d_in[i];
    const float* cls_sub_w  = (const float*)d_in[5];
    const float* cls_sub_b  = (const float*)d_in[6];
    const float* bbox_sub_w = (const float*)d_in[7];
    const float* bbox_sub_b = (const float*)d_in[8];
    const float* cls_score_w = (const float*)d_in[9];
    const float* cls_score_b = (const float*)d_in[10];
    const float* bbox_pred_w = (const float*)d_in[11];
    const float* bbox_pred_b = (const float*)d_in[12];
    const float* ancs[5];
    for (int i = 0; i < 5; ++i) ancs[i] = (const float*)d_in[13 + i];

    // workspace carve (bf16 stored as ushort)
    unsigned char* ws = (unsigned char*)d_ws;
    size_t off = 0;
    auto carve = [&](size_t bytes) -> void* {
        void* p = ws + off;
        off += (bytes + 255) & ~(size_t)255;
        return p;
    };
    const int MAXHW = 15200;
    unsigned short* w_cls = (unsigned short*)carve((size_t)4 * 256 * KTOT * 2);
    unsigned short* w_box = (unsigned short*)carve((size_t)4 * 256 * KTOT * 2);
    unsigned short* w_scr = (unsigned short*)carve((size_t)768 * KTOT * 2);   // 720 -> 768
    unsigned short* w_prd = (unsigned short*)carve((size_t)64 * KTOT * 2);    // 36 -> 64
    unsigned short* zbuf  = (unsigned short*)carve(512);                      // zero guard
    unsigned short* act0 = (unsigned short*)carve((size_t)N_IMG * MAXHW * C_IN * 2);
    unsigned short* act1 = (unsigned short*)carve((size_t)N_IMG * MAXHW * C_IN * 2);
    unsigned short* act2 = (unsigned short*)carve((size_t)N_IMG * MAXHW * C_IN * 2);
    float* clsbuf = (float*)carve((size_t)N_IMG * MAXHW * 720 * 4);
    float* regbuf = (float*)carve((size_t)N_IMG * MAXHW * 36 * 4);
    float* candB = (float*)carve((size_t)N_IMG * TOT_CAND * 4 * 4);
    float* candS = (float*)carve((size_t)N_IMG * TOT_CAND * 4);
    int*   candC = (int*)carve((size_t)N_IMG * TOT_CAND * 4);
    int*   tmpA = (int*)carve((size_t)N_IMG * 1000 * 4);
    int*   tmpC = (int*)carve((size_t)N_IMG * 1000 * 4);
    float* tmpL = (float*)carve((size_t)N_IMG * 1000 * 4);
    unsigned int* hist  = (unsigned int*)carve((size_t)N_IMG * 256 * 4);
    unsigned int* state = (unsigned int*)carve((size_t)N_IMG * 4 * 4);

    // --- weight repacks + zero guard (once per call; cheap)
    repack_w<<<1024, 256, 0, stream>>>(cls_sub_w, w_cls, 4 * 256, 4 * 256);
    repack_w<<<1024, 256, 0, stream>>>(bbox_sub_w, w_box, 4 * 256, 4 * 256);
    repack_w<<<1024, 256, 0, stream>>>(cls_score_w, w_scr, 720, 768);
    repack_w<<<256, 256, 0, stream>>>(bbox_pred_w, w_prd, 36, 64);
    zero_u32<<<1, 128, 0, stream>>>((unsigned int*)zbuf, 128);

    static const int shifts[4] = {24, 16, 8, 0};

    for (int lv = 0; lv < 5; ++lv) {
        const int H = Hs[lv], W = Ws[lv], HW = H * W;
        const int M = HW * A_ANCH * KCLS;
        const int ntk = NTK[lv];

        nchw2nhwc<<<2048, 256, 0, stream>>>(feats[lv], act0, HW);

        dim3 gconv((HW + 63) / 64, 4, N_IMG);
        // --- cls subnet chain: act0 -> act1 -> act2 -> act1 -> act2
        conv3_wmma<true, true><<<gconv, 256, 0, stream>>>(act0, w_cls + 0 * 256 * KTOT, cls_sub_b + 0, zbuf, act1, H, W, 256);
        conv3_wmma<true, true><<<gconv, 256, 0, stream>>>(act1, w_cls + 1 * 256 * KTOT, cls_sub_b + 256, zbuf, act2, H, W, 256);
        conv3_wmma<true, true><<<gconv, 256, 0, stream>>>(act2, w_cls + 2 * 256 * KTOT, cls_sub_b + 512, zbuf, act1, H, W, 256);
        conv3_wmma<true, true><<<gconv, 256, 0, stream>>>(act1, w_cls + 3 * 256 * KTOT, cls_sub_b + 768, zbuf, act2, H, W, 256);
        dim3 gcls((HW + 63) / 64, 12, N_IMG);  // ceil(768/64)=12
        conv3_wmma<false, false><<<gcls, 256, 0, stream>>>(act2, w_scr, cls_score_b, zbuf, clsbuf, H, W, 720);

        // --- bbox subnet chain (act0 preserved)
        conv3_wmma<true, true><<<gconv, 256, 0, stream>>>(act0, w_box + 0 * 256 * KTOT, bbox_sub_b + 0, zbuf, act1, H, W, 256);
        conv3_wmma<true, true><<<gconv, 256, 0, stream>>>(act1, w_box + 1 * 256 * KTOT, bbox_sub_b + 256, zbuf, act2, H, W, 256);
        conv3_wmma<true, true><<<gconv, 256, 0, stream>>>(act2, w_box + 2 * 256 * KTOT, bbox_sub_b + 512, zbuf, act1, H, W, 256);
        conv3_wmma<true, true><<<gconv, 256, 0, stream>>>(act1, w_box + 3 * 256 * KTOT, bbox_sub_b + 768, zbuf, act2, H, W, 256);
        dim3 greg((HW + 63) / 64, 1, N_IMG);
        conv3_wmma<false, false><<<greg, 256, 0, stream>>>(act2, w_prd, bbox_pred_b, zbuf, regbuf, H, W, 36);

        // --- radix top-k on logits (monotone in sigmoid)
        topk_init<<<2, 256, 0, stream>>>(hist, state, ntk);
        for (int p = 0; p < 4; ++p) {
            topk_hist<<<dim3(512, 1, N_IMG), 256, 0, stream>>>(clsbuf, M, hist, state, p, shifts[p]);
            topk_pick<<<N_IMG, 256, 0, stream>>>(hist, state, shifts[p]);
        }
        topk_compact<<<dim3(512, 1, N_IMG), 256, 0, stream>>>(clsbuf, M, state, tmpA, tmpC, tmpL, ntk);
        decode_kernel<<<dim3((ntk + 255) / 256, 1, N_IMG), 256, 0, stream>>>(
            tmpA, tmpC, tmpL, regbuf, ancs[lv], candB, candS, candC, ntk, LOFF[lv], HW);
    }

    nms_kernel<<<N_IMG, 256, 0, stream>>>(candB, candS, candC, (float*)d_out);
}
